// PairEmbed_76708115906774
// MI455X (gfx1250) — compile-verified, matrix operand order checked by reading, not run on hardware
//
#include <hip/hip_runtime.h>
#include <hip/hip_bf16.h>

typedef __attribute__((ext_vector_type(16))) _Float16 v16h;
typedef __attribute__((ext_vector_type(8)))  _Float16 v8h;
typedef __attribute__((ext_vector_type(4)))  _Float16 v4h;
typedef __attribute__((ext_vector_type(8)))  float    v8f;

#define NUM_ELEMENTS 100
#define EMBED_DIM    256
#define HIDDEN_DIM   512
#define NUM_HEADS    8
#define NUM_GAUSS    50
#define KPAD         320   // 256 emb + 64 padded rbf
#define GPAD         64    // rbf K padded 50 -> 64
#define XSTRIDE      328   // LDS X row stride in halves (16B aligned rows)
#define HSTRIDE      40    // per-wave h buffer row stride in halves
#define TILE_E       64    // edges per block
#define WAVES        4

// ---------------- weight prep: f32 -> f16, padded row-major layouts ----------------
__global__ void prep_weights_kernel(const float* __restrict__ W_in,
                                    const float* __restrict__ gate_W,
                                    const float* __restrict__ W_out,
                                    _Float16* __restrict__ winh,    // [512][320]
                                    _Float16* __restrict__ gateh,   // [512][64]
                                    _Float16* __restrict__ wouth) { // [16][512]
    int i = blockIdx.x * blockDim.x + threadIdx.x;
    if (i < HIDDEN_DIM * KPAD) {
        int r = i / KPAD, c = i - r * KPAD;
        winh[i] = (_Float16)(c < (EMBED_DIM + NUM_GAUSS)
                                 ? W_in[r * (EMBED_DIM + NUM_GAUSS) + c] : 0.0f);
    }
    if (i < HIDDEN_DIM * GPAD) {
        int r = i >> 6, c = i & 63;
        gateh[i] = (_Float16)(c < NUM_GAUSS ? gate_W[r * NUM_GAUSS + c] : 0.0f);
    }
    if (i < 16 * HIDDEN_DIM) {
        int r = i >> 9, c = i & 511;
        wouth[i] = (_Float16)(r < NUM_HEADS ? W_out[r * HIDDEN_DIM + c] : 0.0f);
    }
}

// Build a 16x32 f16 WMMA A/B fragment (v16h per lane) from a row-major matrix.
// Caller passes p = &M[row_or_col][kt*32 + half*8]; ISA layout: VGPRs 0-3 hold
// K = half*8+0..7, VGPRs 4-7 hold K = 16+half*8+0..7.
__device__ __forceinline__ v16h load_frag16(const _Float16* p) {
    v8h lo = *(const v8h*)(p);
    v8h hi = *(const v8h*)(p + 16);
    return __builtin_shufflevector(lo, hi, 0,1,2,3,4,5,6,7,8,9,10,11,12,13,14,15);
}

__global__ __launch_bounds__(128)
void pair_embed_kernel(const int* __restrict__ anum,
                       const int* __restrict__ ei,      // [2,E] flattened
                       const int* __restrict__ e2s,
                       const float* __restrict__ dist,
                       const float* __restrict__ emb,   // [100*100, 256]
                       const float* __restrict__ b_in,  // [512]
                       const float* __restrict__ b_out, // [8]
                       const _Float16* __restrict__ winh,
                       const _Float16* __restrict__ gateh,
                       const _Float16* __restrict__ wouth,
                       float* __restrict__ out,         // [8, E]
                       int E) {
    __shared__ __align__(16) _Float16 sX[TILE_E * XSTRIDE];        // 64 x 320 (padded)
    __shared__ __align__(16) _Float16 sH[WAVES * 16 * HSTRIDE];    // per-wave 16x32 h tile

    const int tid = threadIdx.x;
    const int e0  = blockIdx.x * TILE_E;

    // ---- Phase A: gather pair embeddings, f32 -> f16 into LDS cols [0,256) ----
    {
        int el  = tid >> 1;           // local edge 0..63
        int seg = tid & 1;            // which 128-col half
        int e = e0 + el; if (e >= E) e = E - 1;
        int s = e2s[e];
        int pr = anum[ei[s]] + NUM_ELEMENTS * anum[ei[E + s]];
        const float4* src = (const float4*)(emb + (size_t)pr * EMBED_DIM + seg * 128);
        _Float16* dst = sX + el * XSTRIDE + seg * 128;
#pragma unroll
        for (int k = 0; k < 32; ++k) {
            float4 v = src[k];
            v4h h = { (_Float16)v.x, (_Float16)v.y, (_Float16)v.z, (_Float16)v.w };
            *(v4h*)(dst + k * 4) = h;
        }
    }
    // ---- Phase B: Gaussian RBF into LDS cols [256,320) (zero pad >= 50) ----
    if (tid < TILE_E) {
        int e = e0 + tid; if (e >= E) e = E - 1;
        float d0 = dist[e];
        const float delta = 12.0f / 49.0f;
        const float coeff = -0.5f / (delta * delta);
        _Float16* dst = sX + tid * XSTRIDE + EMBED_DIM;
#pragma unroll
        for (int g = 0; g < GPAD; ++g) {
            float val = 0.0f;
            if (g < NUM_GAUSS) { float d = d0 - (float)g * delta; val = __expf(coeff * d * d); }
            dst[g] = (_Float16)val;
        }
    }
    __syncthreads();

    const int lane = tid & 31;
    const int w    = tid >> 5;        // wave id: owns edges [w*16, w*16+16)
    const int half = lane >> 4;
    const int col  = lane & 15;

    // ---- Preload this wave's A fragments once: 16 edges x 320 K (10 frags) ----
    v16h afrag[10];
    {
        const _Float16* arow = sX + (w * 16 + col) * XSTRIDE + half * 8;
#pragma unroll
        for (int kt = 0; kt < 10; ++kt) afrag[kt] = load_frag16(arow + kt * 32);
    }

    _Float16* hbuf = sH + w * 16 * HSTRIDE;
    v8f attC = {};

    for (int nt = 0; nt < 32; ++nt) {
        const int n = nt * 16 + col;  // this lane's hidden-unit column

        // GEMM1: pre = x @ W_in^T   (K = 320)
        v8f c = {};
        const _Float16* wrow = winh + n * KPAD + half * 8;
#pragma unroll
        for (int kt = 0; kt < 10; ++kt)
            c = __builtin_amdgcn_wmma_f32_16x16x32_f16(
                    false, afrag[kt], false, load_frag16(wrow + kt * 32),
                    (short)0, c, false, false);

        // GEMM2: gate = rbf @ gate_W^T   (K = 64, rbf = frags 8..9)
        v8f g = {};
        const _Float16* grow = gateh + n * GPAD + half * 8;
        g = __builtin_amdgcn_wmma_f32_16x16x32_f16(
                false, afrag[8], false, load_frag16(grow), (short)0, g, false, false);
        g = __builtin_amdgcn_wmma_f32_16x16x32_f16(
                false, afrag[9], false, load_frag16(grow + 32), (short)0, g, false, false);

        // h = silu(pre + b_in) * gate -> f16 into per-wave LDS tile
        const float b = b_in[n];
#pragma unroll
        for (int r = 0; r < 8; ++r) {
            float pre  = c[r] + b;
            float hval = (pre / (1.0f + __expf(-pre))) * g[r];
            hbuf[(r + 8 * half) * HSTRIDE + (nt & 1) * 16 + col] = (_Float16)hval;
        }

        // GEMM3 every 2 n-tiles: att += h[16x32] @ W_out^T slice
        if (nt & 1) {
            const _Float16* hp = hbuf + col * HSTRIDE + half * 8;
            v16h ha = load_frag16(hp);
            const _Float16* wo = wouth + col * HIDDEN_DIM + (nt >> 1) * 32 + half * 8;
            attC = __builtin_amdgcn_wmma_f32_16x16x32_f16(
                       false, ha, false, load_frag16(wo), (short)0, attC, false, false);
        }
    }

    // ---- Epilogue: out[head*E + e] = att + b_out[head] ----
    if (col < NUM_HEADS) {
        const float bo = b_out[col];
#pragma unroll
        for (int r = 0; r < 8; ++r) {
            int e = e0 + w * 16 + r + 8 * half;
            if (e < E) out[(size_t)col * E + e] = attC[r] + bo;
        }
    }
}

extern "C" void kernel_launch(void* const* d_in, const int* in_sizes, int n_in,
                              void* d_out, int out_size, void* d_ws, size_t ws_size,
                              hipStream_t stream) {
    const int*   anum  = (const int*)d_in[0];
    const int*   ei    = (const int*)d_in[1];
    const int*   e2s   = (const int*)d_in[2];
    const float* dist  = (const float*)d_in[3];
    const float* emb   = (const float*)d_in[4];
    const float* gateW = (const float*)d_in[5];
    const float* W_in  = (const float*)d_in[6];
    const float* b_in  = (const float*)d_in[7];
    const float* W_out = (const float*)d_in[8];
    const float* b_out = (const float*)d_in[9];
    float* out = (float*)d_out;
    const int E = in_sizes[3];

    // workspace: f16 weights (409,600 bytes total)
    _Float16* winh  = (_Float16*)d_ws;
    _Float16* gateh = winh + HIDDEN_DIM * KPAD;
    _Float16* wouth = gateh + HIDDEN_DIM * GPAD;

    prep_weights_kernel<<<(HIDDEN_DIM * KPAD + 255) / 256, 256, 0, stream>>>(
        W_in, gateW, W_out, winh, gateh, wouth);

    const int nblocks = (E + TILE_E - 1) / TILE_E;
    pair_embed_kernel<<<nblocks, 128, 0, stream>>>(
        anum, ei, e2s, dist, emb, b_in, b_out, winh, gateh, wouth, out, E);
}
